// RouteNet_Fermi_11922829213852
// MI455X (gfx1250) — matrix operand order repaired and checked by reading, not compiled, not run on hardware
//
#include <hip/hip_runtime.h>
#include <hip/hip_bf16.h>
#include <cstdint>
#include <cstddef>

// ---------------------------------------------------------------------------
// RouteNet-Fermi forward on MI455X (gfx1250).
//
// Strategy (compile-only reasoning):
//  * ~45 GFLOPs total, working set < 90 MB -> resident in 192MB L2; the
//    problem is sequential-dependency bound, so we use f16 WMMA
//    (v_wmma_f32_16x16x32_f16, fp32 accumulate) for all LSTM GEMMs and keep
//    recurrent state in registers: one wave32 owns 16 rows and loops time.
//  * h recurrence needs a D-layout -> A-layout transpose each step: done via
//    a per-wave LDS slab (single-wave workgroups; barrier == S_NOP).
//  * Branch-free sigmoid/tanh (v_exp_f32 + v_rcp_f32) so gate math co-issues
//    with the WMMA (XDL) pipe instead of exec-mask branching (ocml tanhf).
//  * 8 message-passing rounds = a short stream of kernels (graph-replayed).
//
// Input order (setup_inputs dict order; params dict flattened as jax pytree
// leaves = alphabetical key order):
//   0 traffic  1 packets  2 eq_lambda  3 avg_pkts_lambda  4 exp_max_factor
//   5 pkts_lambda_on  6 avg_t_off  7 avg_t_on  8 ar_a  9 sigma
//  10 capacity  11 queue_size  12 weight  13 model  14 policy  15 priority
//  16 length  17 queue_to_path  18 link_to_path  19 path_to_link
//  20 path_to_queue  21 queue_to_link
//  23 bw_Wh 24 bw_Wx 25 bw_b 26 fw_Wh 27 fw_Wx 28 fw_b
//  29 le_W1 30 le_W2 31 le_b1 32 le_b2 33 lk_Wh 34 lk_Wx 35 lk_b
//  36 pe_W1 37 pe_W2 38 pe_b1 39 pe_b2 40 qe_W1 41 qe_W2 42 qe_b1 43 qe_b2
//  44 qu_Wh 45 qu_Wx 46 qu_b 47 ro_W1 48 ro_W2 49 ro_W3 50 ro_b1 51 ro_b2
//  52 ro_b3
// ---------------------------------------------------------------------------

#define NPATHS  32768
#define LSEQ    8
#define NLINKS  4096
#define NQUEUES 12288
#define PLn     64
#define PQn     32
#define QPLn    3
#define DDIM    32
#define HDIM    16

typedef __attribute__((ext_vector_type(16))) _Float16 v16h;
typedef __attribute__((ext_vector_type(8)))  float    v8f;

static __device__ __forceinline__ v8f wmma16(v16h a, v16h b, v8f c) {
  // D = A(16x32 f16) * B(32x16 f16) + C(16x16 f32)
  return __builtin_amdgcn_wmma_f32_16x16x32_f16(false, a, false, b, (short)0, c,
                                                false, false);
}

// Branch-free activations: single v_exp_f32 + v_rcp_f32 each, no exec-mask
// manipulation (ocml tanhf is branchy and breaks up the WMMA clusters).
static __device__ __forceinline__ float sigm(float x) {
  return __builtin_amdgcn_rcpf(1.0f + __expf(-x));
}
static __device__ __forceinline__ float fast_tanh(float x) {
  // tanh(x) = 1 - 2/(exp(2x)+1); saturates cleanly (exp->0 => -1, exp->inf => 1)
  const float e = __expf(2.0f * x);
  return 1.0f - 2.0f * __builtin_amdgcn_rcpf(e + 1.0f);
}

static __device__ __forceinline__ v8f splat8(float x) {
  v8f r;
#pragma unroll
  for (int i = 0; i < 8; ++i) r[i] = x;
  return r;
}

// B fragment (32x16, KxN) from row-major f16 W[ld] at (k0,n0).
// Layout: lane n=l&15, kb=(l>=16?8:0); halves 0..7 -> K=kb+i,
// halves 8..15 -> K=16+kb+i.
static __device__ __forceinline__ v16h ldB(const _Float16* W, int ldw, int k0,
                                           int n0) {
  const int l = threadIdx.x & 31;
  const _Float16* p = W + (size_t)(k0 + ((l >> 4) << 3)) * ldw + n0 + (l & 15);
  v16h b;
#pragma unroll
  for (int i = 0; i < 8; ++i) { b[i] = p[(size_t)i * ldw]; b[8 + i] = p[(size_t)(16 + i) * ldw]; }
  return b;
}

// A fragment (16x32, MxK) from row-major f16 X[ld] at (m0,k0).
// Layout: lane m=l&15, kb=(l>=16?8:0); halves 0..7 -> K=kb+i,
// halves 8..15 -> K=16+kb+i.
static __device__ __forceinline__ v16h ldA(const _Float16* X, int ldx, int m0,
                                           int k0) {
  const int l = threadIdx.x & 31;
  const int kb = (l >> 4) << 3;
  const _Float16* p = X + (size_t)(m0 + (l & 15)) * ldx + k0;
  v16h a;
#pragma unroll
  for (int i = 0; i < 8; ++i) { a[i] = p[kb + i]; a[8 + i] = p[16 + kb + i]; }
  return a;
}

// A fragment from fp32 row-major, full K=32 (convert to f16).
static __device__ __forceinline__ v16h ldA_f32(const float* X, int ldx, int m0) {
  const int l = threadIdx.x & 31;
  const int kb = (l >> 4) << 3;
  const float* p = X + (size_t)(m0 + (l & 15)) * ldx;
  v16h a;
#pragma unroll
  for (int i = 0; i < 8; ++i) {
    a[i] = (_Float16)p[kb + i];
    a[8 + i] = (_Float16)p[16 + kb + i];
  }
  return a;
}

// A fragment from fp32 [row][16], K=16 zero-padded to 32 (for HD=16 hidden).
static __device__ __forceinline__ v16h ldA_f32_pad(const float* X, int m0) {
  const int l = threadIdx.x & 31;
  const int kb = (l >> 4) << 3;
  const float* p = X + (size_t)(m0 + (l & 15)) * HDIM;
  v16h a;
#pragma unroll
  for (int i = 0; i < 8; ++i) { a[i] = (_Float16)p[kb + i]; a[8 + i] = (_Float16)0.f; }
  return a;
}

// A fragment from LDS fp32 [16][16], K padded to 32.
static __device__ __forceinline__ v16h ldA_lds_pad16(const float* s) {
  const int l = threadIdx.x & 31;
  const int m = l & 15;
  const int kb = (l >> 4) << 3;
  v16h a;
#pragma unroll
  for (int i = 0; i < 8; ++i) {
    a[i] = (_Float16)s[m * 16 + kb + i];
    a[8 + i] = (_Float16)0.f;
  }
  return a;
}

// A fragment from LDS fp32 [16][32], full K=32.
static __device__ __forceinline__ v16h ldA_lds32(const float* s) {
  const int l = threadIdx.x & 31;
  const int m = l & 15;
  const int kb = (l >> 4) << 3;
  v16h a;
#pragma unroll
  for (int i = 0; i < 8; ++i) {
    a[i] = (_Float16)s[m * 32 + kb + i];
    a[8 + i] = (_Float16)s[m * 32 + 16 + kb + i];
  }
  return a;
}

// ---------------------------------------------------------------------------
// Weight prep: fp32 -> f16 packed block (Wh for bi-LSTM zero-padded K 16->32).
// wf16 layout (halves): fwWx@0(4096) bwWx@4096 fwWh@8192(2048,pad)
// bwWh@10240 quWx@12288(4096) quWh@16384 lkWx@20480 lkWh@24576  (28672 total)
// ---------------------------------------------------------------------------
__global__ void k_prep_w(const float* __restrict__ fwWx,
                         const float* __restrict__ bwWx,
                         const float* __restrict__ fwWh,
                         const float* __restrict__ bwWh,
                         const float* __restrict__ quWx,
                         const float* __restrict__ quWh,
                         const float* __restrict__ lkWx,
                         const float* __restrict__ lkWh,
                         _Float16* __restrict__ w) {
  const int i = blockIdx.x * blockDim.x + threadIdx.x;
  if (i >= 28672) return;
  float v;
  if (i < 4096) {
    v = fwWx[i];
  } else if (i < 8192) {
    v = bwWx[i - 4096];
  } else if (i < 10240) {
    int r = (i - 8192) >> 6, c = (i - 8192) & 63;
    v = (r < HDIM) ? fwWh[r * 64 + c] : 0.f;
  } else if (i < 12288) {
    int r = (i - 10240) >> 6, c = (i - 10240) & 63;
    v = (r < HDIM) ? bwWh[r * 64 + c] : 0.f;
  } else if (i < 16384) {
    v = quWx[i - 12288];
  } else if (i < 20480) {
    v = quWh[i - 16384];
  } else if (i < 24576) {
    v = lkWx[i - 20480];
  } else {
    v = lkWh[i - 24576];
  }
  w[i] = (_Float16)v;
}

// ---------------------------------------------------------------------------
// Embedding MLPs (tiny; VALU with weights staged in LDS).
// ---------------------------------------------------------------------------
__global__ void k_path_embed(
    const float* __restrict__ traffic, const float* __restrict__ packets,
    const float* __restrict__ eq_lambda, const float* __restrict__ avg_pkts_lambda,
    const float* __restrict__ exp_max_factor, const float* __restrict__ pkts_lambda_on,
    const float* __restrict__ avg_t_off, const float* __restrict__ avg_t_on,
    const float* __restrict__ ar_a, const float* __restrict__ sigma,
    const int* __restrict__ model,
    const float* __restrict__ W1, const float* __restrict__ b1,
    const float* __restrict__ W2, const float* __restrict__ b2,
    float* __restrict__ h_fw, float* __restrict__ h_bw) {
  __shared__ float sW1[17 * 32], sb1[32], sW2[32 * 32], sb2[32];
  for (int i = threadIdx.x; i < 17 * 32; i += blockDim.x) sW1[i] = W1[i];
  for (int i = threadIdx.x; i < 32 * 32; i += blockDim.x) sW2[i] = W2[i];
  for (int i = threadIdx.x; i < 32; i += blockDim.x) { sb1[i] = b1[i]; sb2[i] = b2[i]; }
  __syncthreads();
  const int p = blockIdx.x * blockDim.x + threadIdx.x;
  if (p >= NPATHS) return;
  float f[17];
  f[0] = (traffic[p] - 1385.4059f) * (1.f / 859.8119f);
  f[1] = (packets[p] - 1.4015f) * (1.f / 0.8933f);
  const int m = model[p];
#pragma unroll
  for (int k = 0; k < 7; ++k) f[2 + k] = (m == k) ? 1.f : 0.f;
  f[9]  = (eq_lambda[p] - 1350.9712f) * (1.f / 858.3162f);
  f[10] = (avg_pkts_lambda[p] - 0.9117f) * (1.f / 0.9724f);
  f[11] = (exp_max_factor[p] - 6.6636f) * (1.f / 4.7151f);
  f[12] = (pkts_lambda_on[p] - 0.9116f) * (1.f / 1.6513f);
  f[13] = (avg_t_off[p] - 1.6649f) * (1.f / 2.3564f);
  f[14] = (avg_t_on[p] - 1.6649f) * (1.f / 2.3564f);
  f[15] = ar_a[p];
  f[16] = sigma[p];
  float h1[32];
  for (int j = 0; j < 32; ++j) {
    float s = sb1[j];
#pragma unroll
    for (int i = 0; i < 17; ++i) s += f[i] * sW1[i * 32 + j];
    h1[j] = fmaxf(s, 0.f);
  }
  for (int j = 0; j < 32; ++j) {
    float s = sb2[j];
#pragma unroll 8
    for (int i = 0; i < 32; ++i) s += h1[i] * sW2[i * 32 + j];
    s = fmaxf(s, 0.f);
    if (j < HDIM) h_fw[(size_t)p * HDIM + j] = s;
    else          h_bw[(size_t)p * HDIM + (j - HDIM)] = s;
  }
}

__global__ void k_link_embed(const float* __restrict__ traffic,
                             const float* __restrict__ capacity,
                             const int* __restrict__ ptl,       // [NLINKS][PL][2]
                             const int* __restrict__ policy,
                             const float* __restrict__ W1, const float* __restrict__ b1,
                             const float* __restrict__ W2, const float* __restrict__ b2,
                             float* __restrict__ lk_h) {
  __shared__ float sW1[5 * 32], sb1[32], sW2[32 * 32], sb2[32];
  for (int i = threadIdx.x; i < 5 * 32; i += blockDim.x) sW1[i] = W1[i];
  for (int i = threadIdx.x; i < 32 * 32; i += blockDim.x) sW2[i] = W2[i];
  for (int i = threadIdx.x; i < 32; i += blockDim.x) { sb1[i] = b1[i]; sb2[i] = b2[i]; }
  __syncthreads();
  const int li = blockIdx.x * blockDim.x + threadIdx.x;
  if (li >= NLINKS) return;
  const int* row = ptl + (size_t)li * PLn * 2;
  float s = 0.f;
  for (int j = 0; j < PLn; ++j) s += traffic[row[2 * j]];
  float f[5];
  f[0] = s / capacity[li];
  const int pol = policy[li];
#pragma unroll
  for (int k = 0; k < 4; ++k) f[1 + k] = (pol == k) ? 1.f : 0.f;
  float h1[32];
  for (int j = 0; j < 32; ++j) {
    float t = sb1[j];
#pragma unroll
    for (int i = 0; i < 5; ++i) t += f[i] * sW1[i * 32 + j];
    h1[j] = fmaxf(t, 0.f);
  }
  for (int j = 0; j < 32; ++j) {
    float t = sb2[j];
#pragma unroll 8
    for (int i = 0; i < 32; ++i) t += h1[i] * sW2[i * 32 + j];
    lk_h[(size_t)li * 32 + j] = fmaxf(t, 0.f);
  }
}

__global__ void k_queue_embed(const float* __restrict__ queue_size,
                              const int* __restrict__ priority,
                              const float* __restrict__ weight,
                              const float* __restrict__ W1, const float* __restrict__ b1,
                              const float* __restrict__ W2, const float* __restrict__ b2,
                              float* __restrict__ q_h) {
  __shared__ float sW1[5 * 32], sb1[32], sW2[32 * 32], sb2[32];
  for (int i = threadIdx.x; i < 5 * 32; i += blockDim.x) sW1[i] = W1[i];
  for (int i = threadIdx.x; i < 32 * 32; i += blockDim.x) sW2[i] = W2[i];
  for (int i = threadIdx.x; i < 32; i += blockDim.x) { sb1[i] = b1[i]; sb2[i] = b2[i]; }
  __syncthreads();
  const int q = blockIdx.x * blockDim.x + threadIdx.x;
  if (q >= NQUEUES) return;
  float f[5];
  f[0] = (queue_size[q] - 30259.1055f) * (1.f / 21410.0957f);
  const int pr = priority[q];
#pragma unroll
  for (int k = 0; k < 3; ++k) f[1 + k] = (pr == k) ? 1.f : 0.f;
  f[4] = weight[q];
  float h1[32];
  for (int j = 0; j < 32; ++j) {
    float t = sb1[j];
#pragma unroll
    for (int i = 0; i < 5; ++i) t += f[i] * sW1[i * 32 + j];
    h1[j] = fmaxf(t, 0.f);
  }
  for (int j = 0; j < 32; ++j) {
    float t = sb2[j];
#pragma unroll 8
    for (int i = 0; i < 32; ++i) t += h1[i] * sW2[i * 32 + j];
    q_h[(size_t)q * 32 + j] = fmaxf(t, 0.f);
  }
}

// ---------------------------------------------------------------------------
// combined[p][t][0:32]=queue_h[qtp], [32:64]=link_h[ltp]   (f16)
// ---------------------------------------------------------------------------
__global__ void k_build_combined(const float* __restrict__ q_h,
                                 const float* __restrict__ lk_h,
                                 const int* __restrict__ qtp,   // [P][L]
                                 const int* __restrict__ ltp,   // [P][L]
                                 _Float16* __restrict__ comb) {
  const size_t gid = (size_t)blockIdx.x * blockDim.x + threadIdx.x;
  const int col = (int)(gid & 63);
  const size_t pt = gid >> 6;  // p*8+t
  float v;
  if (col < 32) v = q_h[(size_t)qtp[pt] * 32 + col];
  else          v = lk_h[(size_t)ltp[pt] * 32 + (col - 32)];
  comb[gid] = (_Float16)v;
}

// path_seq slot 0 = concat(h_fw, h_bw) (prev path state)
__global__ void k_seq0(const float* __restrict__ h_fw,
                       const float* __restrict__ h_bw,
                       float* __restrict__ path_seq) {
  const size_t gid = (size_t)blockIdx.x * blockDim.x + threadIdx.x;
  const int c = (int)(gid & 31);
  const size_t p = gid >> 5;
  float v = (c < HDIM) ? h_fw[p * HDIM + c] : h_bw[p * HDIM + (c - HDIM)];
  path_seq[(p * 9) * DDIM + c] = v;
}

// ---------------------------------------------------------------------------
// Bidirectional path LSTM: one wave owns 16 paths for all 8 timesteps.
// z(16x64) = x(16x64)@Wx + h(16x16 pad32)@Wh + b ; 4 gate tiles -> 12 WMMA/step.
// blockIdx.y = direction (0=fw, 1=bw).
// ---------------------------------------------------------------------------
__global__ __launch_bounds__(32) void k_path_lstm(
    const _Float16* __restrict__ comb,      // [P][8][64] f16
    const _Float16* __restrict__ wf16,      // packed weights
    const float* __restrict__ b_fw, const float* __restrict__ b_bw,
    float* __restrict__ h_fw, float* __restrict__ h_bw,
    float* __restrict__ c_fw, float* __restrict__ c_bw,
    float* __restrict__ path_seq) {        // [P][9][32] f32
  __shared__ float xp[16 * 16];
  const int dir = blockIdx.y;
  const int base = blockIdx.x * 16;
  const int l = threadIdx.x;
  const int N = l & 15, hi = l >> 4;
  const _Float16* Wx = wf16 + (size_t)dir * 4096;
  const _Float16* Wh = wf16 + 8192 + (size_t)dir * 2048;  // [32][64], padded
  const float* bg = dir ? b_bw : b_fw;
  float* h_st = dir ? h_bw : h_fw;
  float* c_st = dir ? c_bw : c_fw;

  v16h BWx0[4], BWx1[4], BWh[4];
  float bb[4];
#pragma unroll
  for (int nt = 0; nt < 4; ++nt) {
    BWx0[nt] = ldB(Wx, 64, 0, nt * 16);
    BWx1[nt] = ldB(Wx, 64, 32, nt * 16);
    BWh[nt]  = ldB(Wh, 64, 0, nt * 16);
    bb[nt] = bg[nt * 16 + N];
  }
  v16h hA = ldA_f32_pad(h_st, base);
  float cc[8], hn[8];
#pragma unroll
  for (int v = 0; v < 8; ++v)
    cc[v] = c_st[(size_t)(base + v + 8 * hi) * HDIM + N];

  for (int s = 0; s < LSEQ; ++s) {
    const int t = dir ? (LSEQ - 1 - s) : s;
    const _Float16* xrow = comb + (size_t)t * 64;     // + path*512 inside ldA
    // stage next step's x row into L2/WGP$ (global_prefetch_b8)
    if (s + 1 < LSEQ) {
      const int tn = dir ? (LSEQ - 2 - s) : (s + 1);
      __builtin_prefetch(comb + (size_t)(base + (l & 15)) * 512 + (size_t)tn * 64, 0, 3);
    }
    v16h xA0 = ldA(xrow, 512, base, 0);
    v16h xA1 = ldA(xrow, 512, base, 32);
    v8f z[4];
#pragma unroll
    for (int nt = 0; nt < 4; ++nt) {
      v8f acc = splat8(bb[nt]);
      acc = wmma16(xA0, BWx0[nt], acc);
      acc = wmma16(xA1, BWx1[nt], acc);
      acc = wmma16(hA, BWh[nt], acc);
      z[nt] = acc;
    }
#pragma unroll
    for (int v = 0; v < 8; ++v) {
      const float ig = sigm(z[0][v]);
      const float fg = sigm(z[1][v]);
      const float gg = fast_tanh(z[2][v]);
      const float og = sigm(z[3][v]);
      cc[v] = fg * cc[v] + ig * gg;
      hn[v] = og * fast_tanh(cc[v]);
    }
    // scatter h_t into path_seq slot 1+t (fw: cols 0..15, bw: cols 16..31)
#pragma unroll
    for (int v = 0; v < 8; ++v)
      path_seq[((size_t)(base + v + 8 * hi) * 9 + 1 + t) * DDIM + dir * HDIM + N] = hn[v];
    // D-layout -> A-layout via per-wave LDS slab
    __syncthreads();
#pragma unroll
    for (int v = 0; v < 8; ++v) xp[(v + 8 * hi) * 16 + N] = hn[v];
    __syncthreads();
    hA = ldA_lds_pad16(xp);
  }
#pragma unroll
  for (int v = 0; v < 8; ++v) {
    const size_t off = (size_t)(base + v + 8 * hi) * HDIM + N;
    h_st[off] = hn[v];
    c_st[off] = cc[v];
  }
}

// ---------------------------------------------------------------------------
// Queue update: path_sum = sum_{j<32} path_seq[ptq[q][j][0], ptq[q][j][1]];
// (h,c) = LSTMCell(qu, path_sum, h, c).  16 queues per wave, 16 WMMA.
// ---------------------------------------------------------------------------
__global__ __launch_bounds__(32) void k_queue_update(
    const float* __restrict__ path_seq,
    const int* __restrict__ ptq,            // [Q][32][2]
    const _Float16* __restrict__ Wx,        // [32][128]
    const _Float16* __restrict__ Wh,        // [32][128]
    const float* __restrict__ b,            // [128]
    float* __restrict__ q_h, float* __restrict__ q_c) {
  __shared__ float psum[16 * 32];
  const int base = blockIdx.x * 16;
  const int l = threadIdx.x;
  const int N = l & 15, hi = l >> 4;
  // lane l accumulates column l of path_sum for all 16 queues in the tile
  for (int m = 0; m < 16; ++m) {
    const int* idx = ptq + (size_t)(base + m) * (PQn * 2);
    float s = 0.f;
    for (int j = 0; j < PQn; ++j) {
      const int p = idx[2 * j];
      const int sl = idx[2 * j + 1];  // 0..8 into path_seq axis 1
      s += path_seq[((size_t)p * 9 + sl) * DDIM + l];
    }
    psum[m * 32 + l] = s;
  }
  __syncthreads();
  v16h psA = ldA_lds32(psum);
  v16h qhA = ldA_f32(q_h, 32, base);
  v8f z[8];
#pragma unroll
  for (int nt = 0; nt < 8; ++nt) {
    v8f acc = splat8(b[nt * 16 + N]);
    acc = wmma16(psA, ldB(Wx, 128, 0, nt * 16), acc);
    acc = wmma16(qhA, ldB(Wh, 128, 0, nt * 16), acc);
    z[nt] = acc;
  }
#pragma unroll
  for (int g2 = 0; g2 < 2; ++g2) {
#pragma unroll
    for (int v = 0; v < 8; ++v) {
      const float ig = sigm(z[g2][v]);
      const float fg = sigm(z[2 + g2][v]);
      const float gg = fast_tanh(z[4 + g2][v]);
      const float og = sigm(z[6 + g2][v]);
      const size_t off = (size_t)(base + v + 8 * hi) * 32 + g2 * 16 + N;
      float c = q_c[off];
      c = fg * c + ig * gg;
      q_c[off] = c;
      q_h[off] = og * fast_tanh(c);
    }
  }
}

// ---------------------------------------------------------------------------
// Link update: 3-step LSTM over queue_h[queue_to_link[l][j]]; keep final h.
// 16 links per wave, 16 WMMA per step, h recycled via LDS transpose.
// ---------------------------------------------------------------------------
__global__ __launch_bounds__(32) void k_link_update(
    const float* __restrict__ q_h,
    const int* __restrict__ qtl,            // [NLINKS][3]
    const _Float16* __restrict__ Wx,        // [32][128]
    const _Float16* __restrict__ Wh,        // [32][128]
    const float* __restrict__ b,            // [128]
    float* __restrict__ lk_h, float* __restrict__ lk_c) {
  __shared__ float hbuf[16 * 32];
  const int base = blockIdx.x * 16;
  const int l = threadIdx.x;
  const int N = l & 15, hi = l >> 4;
  v16h BWx[8], BWh[8];
  float bb[8];
#pragma unroll
  for (int nt = 0; nt < 8; ++nt) {
    BWx[nt] = ldB(Wx, 128, 0, nt * 16);
    BWh[nt] = ldB(Wh, 128, 0, nt * 16);
    bb[nt] = b[nt * 16 + N];
  }
  v16h hA = ldA_f32(lk_h, 32, base);
  float cc[2][8], hn[2][8];
#pragma unroll
  for (int g2 = 0; g2 < 2; ++g2)
#pragma unroll
    for (int v = 0; v < 8; ++v)
      cc[g2][v] = lk_c[(size_t)(base + v + 8 * hi) * 32 + g2 * 16 + N];

  for (int j = 0; j < QPLn; ++j) {
    // x A-fragment: lane row m -> link base+m -> queue qtl[link][j]
    const int q = qtl[(size_t)(base + (l & 15)) * QPLn + j];
    const float* qp = q_h + (size_t)q * 32;
    const int kb = hi << 3;
    v16h xA;
#pragma unroll
    for (int i = 0; i < 8; ++i) {
      xA[i] = (_Float16)qp[kb + i];
      xA[8 + i] = (_Float16)qp[16 + kb + i];
    }
    v8f z[8];
#pragma unroll
    for (int nt = 0; nt < 8; ++nt) {
      v8f acc = splat8(bb[nt]);
      acc = wmma16(xA, BWx[nt], acc);
      acc = wmma16(hA, BWh[nt], acc);
      z[nt] = acc;
    }
#pragma unroll
    for (int g2 = 0; g2 < 2; ++g2)
#pragma unroll
      for (int v = 0; v < 8; ++v) {
        const float ig = sigm(z[g2][v]);
        const float fg = sigm(z[2 + g2][v]);
        const float gg = fast_tanh(z[4 + g2][v]);
        const float og = sigm(z[6 + g2][v]);
        cc[g2][v] = fg * cc[g2][v] + ig * gg;
        hn[g2][v] = og * fast_tanh(cc[g2][v]);
      }
    __syncthreads();
#pragma unroll
    for (int g2 = 0; g2 < 2; ++g2)
#pragma unroll
      for (int v = 0; v < 8; ++v)
        hbuf[(v + 8 * hi) * 32 + g2 * 16 + N] = hn[g2][v];
    __syncthreads();
    hA = ldA_lds32(hbuf);
  }
#pragma unroll
  for (int g2 = 0; g2 < 2; ++g2)
#pragma unroll
    for (int v = 0; v < 8; ++v) {
      const size_t off = (size_t)(base + v + 8 * hi) * 32 + g2 * 16 + N;
      lk_h[off] = hn[g2][v];
      lk_c[off] = cc[g2][v];
    }
}

// ---------------------------------------------------------------------------
// Readout: ratio = MLP3(path_seq[:,1:]); jitter = sum_t ratio/cap[ltp].
// ---------------------------------------------------------------------------
__global__ void k_readout(const float* __restrict__ path_seq,
                          const int* __restrict__ ltp,       // [P][L]
                          const float* __restrict__ capacity,
                          const float* __restrict__ W1, const float* __restrict__ W2,
                          const float* __restrict__ W3, const float* __restrict__ b1,
                          const float* __restrict__ b2, const float* __restrict__ b3,
                          float* __restrict__ out) {
  __shared__ float sW1[32 * 16], sW2[16 * 16], sW3[16], sb1[16], sb2[16];
  for (int i = threadIdx.x; i < 32 * 16; i += blockDim.x) sW1[i] = W1[i];
  for (int i = threadIdx.x; i < 16 * 16; i += blockDim.x) sW2[i] = W2[i];
  for (int i = threadIdx.x; i < 16; i += blockDim.x) {
    sW3[i] = W3[i]; sb1[i] = b1[i]; sb2[i] = b2[i];
  }
  __syncthreads();
  const int p = blockIdx.x * blockDim.x + threadIdx.x;
  if (p >= NPATHS) return;
  const float b3v = b3[0];
  float acc = 0.f;
  for (int t = 0; t < LSEQ; ++t) {
    const float* x = path_seq + ((size_t)p * 9 + 1 + t) * DDIM;
    float h1[16];
    for (int j = 0; j < 16; ++j) {
      float s = sb1[j];
#pragma unroll 8
      for (int i = 0; i < 32; ++i) s += x[i] * sW1[i * 16 + j];
      h1[j] = fmaxf(s, 0.f);
    }
    float h2[16];
    for (int j = 0; j < 16; ++j) {
      float s = sb2[j];
#pragma unroll
      for (int i = 0; i < 16; ++i) s += h1[i] * sW2[i * 16 + j];
      h2[j] = fmaxf(s, 0.f);
    }
    float r = b3v;
#pragma unroll
    for (int i = 0; i < 16; ++i) r += h2[i] * sW3[i];
    acc += r / capacity[ltp[(size_t)p * LSEQ + t]];
  }
  out[p] = acc;
}

// ---------------------------------------------------------------------------
// Host orchestration.
// ---------------------------------------------------------------------------
extern "C" void kernel_launch(void* const* d_in, const int* in_sizes, int n_in,
                              void* d_out, int out_size, void* d_ws, size_t ws_size,
                              hipStream_t stream) {
  (void)in_sizes; (void)n_in; (void)out_size; (void)ws_size;
  // ---- inputs ----
  const float* traffic        = (const float*)d_in[0];
  const float* packets        = (const float*)d_in[1];
  const float* eq_lambda      = (const float*)d_in[2];
  const float* avg_pkts_lambda= (const float*)d_in[3];
  const float* exp_max_factor = (const float*)d_in[4];
  const float* pkts_lambda_on = (const float*)d_in[5];
  const float* avg_t_off      = (const float*)d_in[6];
  const float* avg_t_on       = (const float*)d_in[7];
  const float* ar_a           = (const float*)d_in[8];
  const float* sigma          = (const float*)d_in[9];
  const float* capacity       = (const float*)d_in[10];
  const float* queue_size     = (const float*)d_in[11];
  const float* weight         = (const float*)d_in[12];
  const int*   model          = (const int*)d_in[13];
  const int*   policy         = (const int*)d_in[14];
  const int*   priority       = (const int*)d_in[15];
  const int*   qtp            = (const int*)d_in[17];
  const int*   ltp            = (const int*)d_in[18];
  const int*   ptl            = (const int*)d_in[19];
  const int*   ptq            = (const int*)d_in[20];
  const int*   qtl            = (const int*)d_in[21];
  // params (pytree/alphabetical order)
  const float* bw_Wh = (const float*)d_in[23];
  const float* bw_Wx = (const float*)d_in[24];
  const float* bw_b  = (const float*)d_in[25];
  const float* fw_Wh = (const float*)d_in[26];
  const float* fw_Wx = (const float*)d_in[27];
  const float* fw_b  = (const float*)d_in[28];
  const float* le_W1 = (const float*)d_in[29];
  const float* le_W2 = (const float*)d_in[30];
  const float* le_b1 = (const float*)d_in[31];
  const float* le_b2 = (const float*)d_in[32];
  const float* lk_Wh = (const float*)d_in[33];
  const float* lk_Wx = (const float*)d_in[34];
  const float* lk_b  = (const float*)d_in[35];
  const float* pe_W1 = (const float*)d_in[36];
  const float* pe_W2 = (const float*)d_in[37];
  const float* pe_b1 = (const float*)d_in[38];
  const float* pe_b2 = (const float*)d_in[39];
  const float* qe_W1 = (const float*)d_in[40];
  const float* qe_W2 = (const float*)d_in[41];
  const float* qe_b1 = (const float*)d_in[42];
  const float* qe_b2 = (const float*)d_in[43];
  const float* qu_Wh = (const float*)d_in[44];
  const float* qu_Wx = (const float*)d_in[45];
  const float* qu_b  = (const float*)d_in[46];
  const float* ro_W1 = (const float*)d_in[47];
  const float* ro_W2 = (const float*)d_in[48];
  const float* ro_W3 = (const float*)d_in[49];
  const float* ro_b1 = (const float*)d_in[50];
  const float* ro_b2 = (const float*)d_in[51];
  const float* ro_b3 = (const float*)d_in[52];

  // ---- workspace carve-up (~84 MB) ----
  char* ws = (char*)d_ws;
  size_t off = 0;
  _Float16* comb = (_Float16*)(ws + off); off += (size_t)NPATHS * LSEQ * 64 * 2;   // 32 MB
  float* path_seq = (float*)(ws + off);   off += (size_t)NPATHS * 9 * DDIM * 4;    // 36 MB
  float* h_fw = (float*)(ws + off);       off += (size_t)NPATHS * HDIM * 4;
  float* h_bw = (float*)(ws + off);       off += (size_t)NPATHS * HDIM * 4;
  float* c_fw = (float*)(ws + off);       off += (size_t)NPATHS * HDIM * 4;
  float* c_bw = (float*)(ws + off);       off += (size_t)NPATHS * HDIM * 4;
  float* q_h  = (float*)(ws + off);       off += (size_t)NQUEUES * 32 * 4;
  float* q_c  = (float*)(ws + off);       off += (size_t)NQUEUES * 32 * 4;
  float* lkh  = (float*)(ws + off);       off += (size_t)NLINKS * 32 * 4;
  float* lkc  = (float*)(ws + off);       off += (size_t)NLINKS * 32 * 4;
  _Float16* wf16 = (_Float16*)(ws + off); off += 28672 * 2;

  // zero LSTM cell states every launch (deterministic)
  hipMemsetAsync(c_fw, 0, (size_t)NPATHS * HDIM * 4, stream);
  hipMemsetAsync(c_bw, 0, (size_t)NPATHS * HDIM * 4, stream);
  hipMemsetAsync(q_c, 0, (size_t)NQUEUES * 32 * 4, stream);
  hipMemsetAsync(lkc, 0, (size_t)NLINKS * 32 * 4, stream);

  // weight prep + embeddings
  k_prep_w<<<112, 256, 0, stream>>>(fw_Wx, bw_Wx, fw_Wh, bw_Wh, qu_Wx, qu_Wh,
                                    lk_Wx, lk_Wh, wf16);
  k_path_embed<<<NPATHS / 256, 256, 0, stream>>>(
      traffic, packets, eq_lambda, avg_pkts_lambda, exp_max_factor,
      pkts_lambda_on, avg_t_off, avg_t_on, ar_a, sigma, model,
      pe_W1, pe_b1, pe_W2, pe_b2, h_fw, h_bw);
  k_link_embed<<<NLINKS / 256, 256, 0, stream>>>(
      traffic, capacity, ptl, policy, le_W1, le_b1, le_W2, le_b2, lkh);
  k_queue_embed<<<NQUEUES / 256, 256, 0, stream>>>(
      queue_size, priority, weight, qe_W1, qe_b1, qe_W2, qe_b2, q_h);

  // 8 message-passing rounds
  for (int it = 0; it < 8; ++it) {
    k_build_combined<<<(NPATHS * LSEQ * 64) / 256, 256, 0, stream>>>(
        q_h, lkh, qtp, ltp, comb);
    k_seq0<<<(NPATHS * DDIM) / 256, 256, 0, stream>>>(h_fw, h_bw, path_seq);
    k_path_lstm<<<dim3(NPATHS / 16, 2), 32, 0, stream>>>(
        comb, wf16, fw_b, bw_b, h_fw, h_bw, c_fw, c_bw, path_seq);
    k_queue_update<<<NQUEUES / 16, 32, 0, stream>>>(
        path_seq, ptq, wf16 + 12288, wf16 + 16384, qu_b, q_h, q_c);
    k_link_update<<<NLINKS / 16, 32, 0, stream>>>(
        q_h, qtl, wf16 + 20480, wf16 + 24576, lk_b, lkh, lkc);
  }

  k_readout<<<NPATHS / 256, 256, 0, stream>>>(
      path_seq, ltp, capacity, ro_W1, ro_W2, ro_W3, ro_b1, ro_b2, ro_b3,
      (float*)d_out);
}
